// GNN_70282844832114
// MI455X (gfx1250) — compile-verified, matrix operand order checked by reading, not run on hardware
//
#include <hip/hip_runtime.h>
#include <hip/hip_bf16.h>
#include <cstdint>

// ---------------------------------------------------------------------------
// Types for CDNA5 WMMA (wave32): v_wmma_f32_16x16x32_f16
// ---------------------------------------------------------------------------
typedef __attribute__((ext_vector_type(16))) _Float16 v16h;
typedef __attribute__((ext_vector_type(8)))  float    v8f;

#define WPT 4            // waves (16-row tiles) per block
#define GEMM_BLOCK 128   // 4 waves * 32 lanes

static __device__ __forceinline__ int kslot(int j, int half) {
    // ISA 16-bit A-matrix 16x32 layout: lanes 0-15 K={0..7,16..23}, lanes 16-31 K={8..15,24..31}
    return (j < 4 ? 2 * j : 16 + 2 * (j - 4)) + 8 * half;
}

// ---------------------------------------------------------------------------
// Generic WMMA GEMM: C[M,N] = act( A * Bw + bias )
//   Bw is f16 [Kpad x N] (pre-transposed, zero-padded weights)
//   MODE 0: A[m,k] = X[m*Kreal + k]
//   MODE 1: A[m,k] = X[dst[m]*Cfeat + k]                          (gather x_i)
//   MODE 2: A[m,k] = k<C ? X[dst[m]*C+k]
//                        : X[src[m]*C+(k-C)] - X[dst[m]*C+(k-C)]  (concat [xi, xj-xi])
//   act: 0 none, 1 relu, 2 tanh
// All A loads are unconditional (clamped addresses + cndmask selects): no
// divergent control flow in the K loop.
// ---------------------------------------------------------------------------
template <int MODE>
__global__ __launch_bounds__(GEMM_BLOCK)
void wmma_gemm_kernel(const float* __restrict__ X,
                      const int* __restrict__ src, const int* __restrict__ dst,
                      const _Float16* __restrict__ Bw, const float* __restrict__ bias,
                      float* __restrict__ C,
                      int M, int N, int Kpad, int Kreal, int Cfeat, int act)
{
    // B staged in *fragment order*: sB[((chunk*32 + lane)*16 + e)] so the inner
    // loop reads one contiguous 32-byte v16h per chunk (2x ds_load_b128).
    __shared__ _Float16 sB[256 * 16];   // Kpad<=256 (8 KB)

    const int tid  = threadIdx.x;
    const int lane = tid & 31;
    const int wave = tid >> 5;
    const int ntile = blockIdx.y;

    for (int i = tid; i < Kpad * 16; i += GEMM_BLOCK) {
        const int chunk = i >> 9;           // 512 elements per K-chunk
        const int r     = i & 511;
        const int l     = r >> 4;           // lane this element belongs to
        const int e     = r & 15;           // element slot within fragment
        const int j     = e >> 1, t = e & 1;
        const int h     = l >> 4, c = l & 15;
        const int k     = chunk * 32 + kslot(j, h) + t;
        sB[i] = Bw[k * N + ntile * 16 + c];
    }
    __syncthreads();

    const int mtile = blockIdx.x * WPT + wave;
    if (mtile * 16 >= M) return;            // no further barriers below

    const int half = lane >> 4;
    const int col  = lane & 15;
    const int mro  = min(mtile * 16 + col, M - 1);   // clamped: garbage rows never stored

    long long baseA0 = 0, baseA1 = 0;
    if (MODE == 0)      baseA0 = (long long)mro * Kreal;
    else if (MODE == 1) baseA0 = (long long)dst[mro] * Cfeat;
    else { baseA0 = (long long)dst[mro] * Cfeat; baseA1 = (long long)src[mro] * Cfeat; }

    const float2* __restrict__ X2 = (const float2*)X;
    const v16h*   __restrict__ sBv = (const v16h*)sB;

    v8f acc = {};
    const int Kfull = Kreal & ~31;          // chunks needing no k-bounds test

    int kc = 0;
    for (; kc < Kfull; kc += 32) {
        v16h a;
#pragma unroll
        for (int j = 0; j < 8; ++j) {
            const int k = kc + kslot(j, half);
            float2 p;
            if (MODE == 2) {
                const bool hi = (k >= Cfeat);
                const int  k2 = hi ? (k - Cfeat) : k;
                const long long i0 = (baseA0 + k2) >> 1;
                const long long i1 = ((hi ? baseA1 : baseA0) + k2) >> 1;
                const float2 v0 = X2[i0];
                const float2 v1 = X2[i1];
                p.x = hi ? (v1.x - v0.x) : v0.x;
                p.y = hi ? (v1.y - v0.y) : v0.y;
            } else {
                p = X2[(baseA0 + k) >> 1];
            }
            a[2 * j]     = (_Float16)p.x;
            a[2 * j + 1] = (_Float16)p.y;
        }
        const v16h b = sBv[(kc >> 5) * 32 + lane];
        acc = __builtin_amdgcn_wmma_f32_16x16x32_f16(false, a, false, b,
                                                     (short)0, acc, false, false);
    }

    if (Kfull < Kpad) {                     // single zero-padded tail chunk
        v16h a;
#pragma unroll
        for (int j = 0; j < 8; ++j) {
#pragma unroll
            for (int t = 0; t < 2; ++t) {
                const int k   = Kfull + kslot(j, half) + t;
                const int kcl = min(k, Kreal - 1);
                float av;
                if (MODE == 2) {
                    const bool hi = (kcl >= Cfeat);
                    const int  k2 = hi ? (kcl - Cfeat) : kcl;
                    const float v0 = X[baseA0 + k2];
                    const float v1 = X[(hi ? baseA1 : baseA0) + k2];
                    av = hi ? (v1 - v0) : v0;
                } else {
                    av = X[baseA0 + kcl];
                }
                av = (k < Kreal) ? av : 0.0f;
                a[2 * j + t] = (_Float16)av;
            }
        }
        const v16h b = sBv[(Kfull >> 5) * 32 + lane];
        acc = __builtin_amdgcn_wmma_f32_16x16x32_f16(false, a, false, b,
                                                     (short)0, acc, false, false);
    }

    const int   n  = ntile * 16 + col;
    const float bv = bias ? bias[n] : 0.0f;
#pragma unroll
    for (int r = 0; r < 8; ++r) {
        const int m = mtile * 16 + r + 8 * half;
        if (m < M) {
            float v = acc[r] + bv;
            if (act == 1)      v = fmaxf(v, 0.0f);
            else if (act == 2) v = tanhf(v);
            C[(long long)m * N + n] = v;
        }
    }
}

// ---------------------------------------------------------------------------
// Weight prep: Bw[k*N + n] = (k < Kin) ? W[n*Kin + k] : 0   (f32 -> f16, pad K)
// ---------------------------------------------------------------------------
__global__ void prep_w_kernel(const float* __restrict__ W, _Float16* __restrict__ Bw,
                              int Kin, int Kpad, int N)
{
    int i = blockIdx.x * blockDim.x + threadIdx.x;
    if (i >= Kpad * N) return;
    int k = i / N, n = i % N;
    Bw[i] = (_Float16)((k < Kin) ? W[n * Kin + k] : 0.0f);
}

__global__ void fill_kernel(float* __restrict__ p, float v, int n)
{
    int i = blockIdx.x * blockDim.x + threadIdx.x;
    if (i < n) p[i] = v;
}

static __device__ void atomicMaxF(float* addr, float val)
{
    int* ia = (int*)addr;
    int old = __hip_atomic_load(ia, __ATOMIC_RELAXED, __HIP_MEMORY_SCOPE_AGENT);
    while (__int_as_float(old) < val) {
        int assumed = old;
        old = atomicCAS(ia, assumed, __float_as_int(val));
        if (old == assumed) break;
    }
}

// w[e] = b7 + sum_j w7[j] * t5[e,j] * t6[e,j]
__global__ void edge_attn_kernel(const float* __restrict__ t5, const float* __restrict__ t6,
                                 const float* __restrict__ w7, const float* __restrict__ b7,
                                 float* __restrict__ we, int E)
{
    int e = blockIdx.x * blockDim.x + threadIdx.x;
    if (e >= E) return;
    float acc = b7[0];
#pragma unroll
    for (int j = 0; j < 16; ++j)
        acc += w7[j] * t5[(long long)e * 16 + j] * t6[(long long)e * 16 + j];
    we[e] = acc;
}

__global__ void seg_max_kernel(const float* __restrict__ w, const int* __restrict__ seg,
                               float* __restrict__ m, int n)
{
    int i = blockIdx.x * blockDim.x + threadIdx.x;
    if (i < n) atomicMaxF(&m[seg[i]], w[i]);
}

__global__ void seg_expsum_kernel(float* __restrict__ w, const int* __restrict__ seg,
                                  const float* __restrict__ m, float* __restrict__ s, int n)
{
    int i = blockIdx.x * blockDim.x + threadIdx.x;
    if (i >= n) return;
    float ex = expf(w[i] - m[seg[i]]);
    w[i] = ex;
    atomicAdd(&s[seg[i]], ex);
}

// mag[dst[e], c] += msg[e, c] * ex[e] / (s[src[e]] + 1e-16)
__global__ void aggregate_kernel(const float* __restrict__ msg, const float* __restrict__ ex,
                                 const float* __restrict__ s,
                                 const int* __restrict__ src, const int* __restrict__ dst,
                                 float* __restrict__ mag, int E, int C)
{
    int i = blockIdx.x * blockDim.x + threadIdx.x;
    if (i >= E * C) return;
    int e = i / C, c = i - e * C;
    float coef = ex[e] / (s[src[e]] + 1e-16f);
    atomicAdd(&mag[(long long)dst[e] * C + c], msg[(long long)e * C + c] * coef);
}

// h = act( sigmoid(mlp3 [xr,m]) * m + sigmoid(mlp4 [xr,m]) * xr )
__global__ void combine_kernel(const float* __restrict__ xr, const float* __restrict__ mag,
                               const float* __restrict__ w3, const float* __restrict__ b3,
                               const float* __restrict__ w4, const float* __restrict__ b4,
                               float* __restrict__ h, int Nn, int C, int relu_out)
{
    int n = blockIdx.x * blockDim.x + threadIdx.x;
    if (n >= Nn) return;
    const float* xrn = xr  + (long long)n * C;
    const float* mn  = mag + (long long)n * C;
    float a3 = b3[0], a4 = b4[0];
    for (int c = 0; c < C; ++c) {
        float xv = xrn[c], mv = mn[c];
        a3 += w3[c] * xv + w3[C + c] * mv;
        a4 += w4[c] * xv + w4[C + c] * mv;
    }
    float g1 = 1.0f / (1.0f + expf(-a3));
    float g2 = 1.0f / (1.0f + expf(-a4));
    for (int c = 0; c < C; ++c) {
        float v = g1 * mn[c] + g2 * xrn[c];
        if (relu_out) v = fmaxf(v, 0.0f);
        h[(long long)n * C + c] = v;
    }
}

__global__ void pool_logits_kernel(const float* __restrict__ h, const float* __restrict__ wg,
                                   const float* __restrict__ bg, float* __restrict__ l,
                                   int Nn, int C)
{
    int n = blockIdx.x * blockDim.x + threadIdx.x;
    if (n >= Nn) return;
    float acc = bg[0];
    for (int c = 0; c < C; ++c) acc += wg[c] * h[(long long)n * C + c];
    l[n] = acc;
}

__global__ void pool_accum_kernel(const float* __restrict__ h, const float* __restrict__ ex,
                                  const float* __restrict__ s, const int* __restrict__ batch,
                                  float* __restrict__ Z, int Nn, int C, int ldz, int coloff)
{
    int i = blockIdx.x * blockDim.x + threadIdx.x;
    if (i >= Nn * C) return;
    int n = i / C, c = i - n * C;
    int g = batch[n];
    float coef = ex[n] / (s[g] + 1e-16f);
    atomicAdd(&Z[(long long)g * ldz + coloff + c], h[(long long)n * C + c] * coef);
}

// out[g] = b + dot(w, z[g,:K])
__global__ void lin_out_kernel(const float* __restrict__ z, const float* __restrict__ w,
                               const float* __restrict__ b, float* __restrict__ out,
                               int G, int K)
{
    int g = blockIdx.x * blockDim.x + threadIdx.x;
    if (g >= G) return;
    float acc = b[0];
    for (int j = 0; j < K; ++j) acc += w[j] * z[(long long)g * K + j];
    out[g] = acc;
}

// ---------------------------------------------------------------------------
// Host orchestration
// ---------------------------------------------------------------------------
extern "C" void kernel_launch(void* const* d_in, const int* in_sizes, int n_in,
                              void* d_out, int out_size, void* d_ws, size_t ws_size,
                              hipStream_t stream)
{
    (void)in_sizes; (void)n_in; (void)out_size; (void)ws_size;

    const int E = 800000, NN = 50000, G = 500;

    const float* x     = (const float*)d_in[0];
    const int*   ei    = (const int*)d_in[1];
    const int*   srcI  = ei;            // edge_index[0]
    const int*   dstI  = ei + E;        // edge_index[1]
    const int*   batch = (const int*)d_in[2];
    auto P = [&](int i) { return (const float*)d_in[i]; };
    // params leaf order (jax tree flatten, alphabetical; each linear = {b, w}):
    // conv1..conv4 (mlp1..mlp7), gate1, gate2, lin1..lin4 starting at d_in[3]

    // ---- workspace bump allocator -------------------------------------------
    char*  ws  = (char*)d_ws;
    size_t off = 0;
    auto alloc = [&](size_t bytes) -> void* {
        void* p = ws + off;
        off = (off + bytes + 255) & ~(size_t)255;
        return p;
    };
    float* MSG  = (float*)alloc((size_t)E * 128 * 4);
    float* T5   = (float*)alloc((size_t)E * 16 * 4);
    float* T6   = (float*)alloc((size_t)E * 16 * 4);
    float* WE   = (float*)alloc((size_t)E * 4);
    float* SEGM = (float*)alloc((size_t)NN * 4);
    float* SEGS = (float*)alloc((size_t)NN * 4);
    float* MAG  = (float*)alloc((size_t)NN * 128 * 4);
    float* XR   = (float*)alloc((size_t)NN * 128 * 4);
    float* HA   = (float*)alloc((size_t)NN * 128 * 4);
    float* HB   = (float*)alloc((size_t)NN * 128 * 4);
    float* NL   = (float*)alloc((size_t)NN * 4);
    float* GSM  = (float*)alloc((size_t)G * 4);
    float* GSS  = (float*)alloc((size_t)G * 4);
    float* Z    = (float*)alloc((size_t)G * 64 * 4);
    float* Z1   = (float*)alloc((size_t)G * 128 * 4);
    float* Z2   = (float*)alloc((size_t)G * 16 * 4);
    float* Z3   = (float*)alloc((size_t)G * 16 * 4);
    char*  BW   = (char*)alloc((size_t)2 << 20);   // f16 weight staging region
    size_t bwoff = 0;

    auto fill = [&](float* p, float v, int n) {
        fill_kernel<<<(n + 255) / 256, 256, 0, stream>>>(p, v, n);
    };

    // GEMM launcher: preps f16 weights then runs the WMMA kernel.
    auto gemm = [&](const float* X, const int* s_, const int* d_,
                    const float* W, const float* b, float* Cout_,
                    int M, int N, int Kreal, int Cfeat, int mode, int act) {
        int Kpad = ((Kreal + 31) / 32) * 32;
        _Float16* Bw = (_Float16*)(BW + bwoff);
        bwoff = (bwoff + (size_t)Kpad * N * 2 + 255) & ~(size_t)255;
        int tot = Kpad * N;
        prep_w_kernel<<<(tot + 255) / 256, 256, 0, stream>>>(W, Bw, Kreal, Kpad, N);
        dim3 grid((M + 16 * WPT - 1) / (16 * WPT), N / 16);
        if (mode == 0)
            wmma_gemm_kernel<0><<<grid, GEMM_BLOCK, 0, stream>>>(
                X, s_, d_, Bw, b, Cout_, M, N, Kpad, Kreal, Cfeat, act);
        else if (mode == 1)
            wmma_gemm_kernel<1><<<grid, GEMM_BLOCK, 0, stream>>>(
                X, s_, d_, Bw, b, Cout_, M, N, Kpad, Kreal, Cfeat, act);
        else
            wmma_gemm_kernel<2><<<grid, GEMM_BLOCK, 0, stream>>>(
                X, s_, d_, Bw, b, Cout_, M, N, Kpad, Kreal, Cfeat, act);
    };

    auto pool = [&](const float* h, const float* wg, const float* bg, int coloff) {
        pool_logits_kernel<<<(NN + 255) / 256, 256, 0, stream>>>(h, wg, bg, NL, NN, 32);
        fill(GSM, -3.0e38f, G);
        fill(GSS, 0.0f, G);
        seg_max_kernel<<<(NN + 255) / 256, 256, 0, stream>>>(NL, batch, GSM, NN);
        seg_expsum_kernel<<<(NN + 255) / 256, 256, 0, stream>>>(NL, batch, GSM, GSS, NN);
        int tot = NN * 32;
        pool_accum_kernel<<<(tot + 255) / 256, 256, 0, stream>>>(
            h, NL, GSS, batch, Z, NN, 32, 64, coloff);
    };

    fill(Z, 0.0f, G * 64);   // pooled features accumulate here (cols 0..31 / 32..63)

    const int cinA[4]  = {4, 128, 32, 64};
    const int coutA[4] = {128, 32, 64, 32};
    const float* hin = x;
    float* hbufs[2] = {HA, HB};

    for (int l = 0; l < 4; ++l) {
        const int base = 3 + l * 14;   // conv-l leaves: mlp-k -> {b: base+2k-2, w: base+2k-1}
        const int ci = cinA[l], co = coutA[l];

        // msg = relu(mlp1 [xi, xj-xi])  — fused edge gather into WMMA A operand
        gemm(hin, srcI, dstI, P(base + 1), P(base + 0), MSG, E, co, 2 * ci, ci, 2, 1);
        // t5 = tanh(mlp5 xi)            — gather-by-dst into WMMA A operand
        gemm(hin, srcI, dstI, P(base + 9), P(base + 8), T5, E, 16, ci, ci, 1, 2);
        // t6 = tanh(mlp6 msg)
        gemm(MSG, nullptr, nullptr, P(base + 11), P(base + 10), T6, E, 16, co, co, 0, 2);
        // w = mlp7(t5 * t6)
        edge_attn_kernel<<<(E + 255) / 256, 256, 0, stream>>>(
            T5, T6, P(base + 13), P(base + 12), WE, E);
        // segment softmax over src
        fill(SEGM, -3.0e38f, NN);
        fill(SEGS, 0.0f, NN);
        seg_max_kernel<<<(E + 255) / 256, 256, 0, stream>>>(WE, srcI, SEGM, E);
        seg_expsum_kernel<<<(E + 255) / 256, 256, 0, stream>>>(WE, srcI, SEGM, SEGS, E);
        // m = scatter-add(msg * w) at dst
        fill(MAG, 0.0f, NN * co);
        {
            int tot = E * co;
            aggregate_kernel<<<(tot + 255) / 256, 256, 0, stream>>>(
                MSG, WE, SEGS, srcI, dstI, MAG, E, co);
        }
        // xr = relu(mlp2 x)  — node GEMM
        gemm(hin, nullptr, nullptr, P(base + 3), P(base + 2), XR, NN, co, ci, ci, 0, 1);
        // gated combine + outer relu
        float* hout = hbufs[l & 1];
        combine_kernel<<<(NN + 255) / 256, 256, 0, stream>>>(
            XR, MAG, P(base + 5), P(base + 4), P(base + 7), P(base + 6), hout, NN, co, 1);

        if (l == 1) pool(hout, P(60), P(59), 0);    // gate1 -> Z[:, 0:32]
        if (l == 3) pool(hout, P(62), P(61), 32);   // gate2 -> Z[:, 32:64]
        hin = hout;
    }

    // ---- head: lin1(relu) -> lin2(relu) -> lin3 -> lin4 ----------------------
    gemm(Z,  nullptr, nullptr, P(64), P(63), Z1, G, 128, 64, 64, 0, 1);     // lin1
    gemm(Z1, nullptr, nullptr, P(66), P(65), Z2, G, 16, 128, 128, 0, 1);    // lin2
    gemm(Z2, nullptr, nullptr, P(68), P(67), Z3, G, 16, 16, 16, 0, 0);      // lin3
    lin_out_kernel<<<(G + 255) / 256, 256, 0, stream>>>(
        Z3, P(70), P(69), (float*)d_out, G, 16);                            // lin4
}
